// SparseSpatialTransformer_75307956568551
// MI455X (gfx1250) — compile-verified, matrix operand order checked by reading, not run on hardware
//
#include <hip/hip_runtime.h>
#include <math.h>

// ---------------------------------------------------------------------------
// Shapes (compile-time constants from the reference)
// ---------------------------------------------------------------------------
#define BB   64
#define TT   16
#define NN   128
#define DD   512
#define EE   512
#define FFD  2048
#define CHN  64
#define BT   (BB*TT)          // 1024
#define KESI 0.3f

typedef __attribute__((ext_vector_type(16))) __bf16 v16bf;
typedef __attribute__((ext_vector_type(8)))  float  v8f;

// ---------------------------------------------------------------------------
// bf16 hi/lo split (a = hi + lo with ~fp32 precision when recombined)
// ---------------------------------------------------------------------------
__device__ __forceinline__ void split_bf16(float x, __bf16& h, __bf16& l) {
    h = (__bf16)x;
    l = (__bf16)(x - (float)h);
}

// Fragment-linear index for a 16x32 bf16 WMMA operand (A: [m][k], B: [k][n]
// with n substituted for m). Returns lane*16 + element so that a lane's 16
// bf16 values are contiguous (32B) in LDS -> ds_load_b128 x2 per fragment.
// Derived from the CDNA5 ISA 16-bit A-matrix 16x32 layout table.
__device__ __forceinline__ int frag_idx(int m16, int k32) {
    const int hi        = k32 >> 4;        // K >= 16 -> VGPR 4..7
    const int within    = k32 & 15;
    const int lane_half = within >> 3;     // lanes 16..31 hold K%16 >= 8
    const int r2        = within & 7;
    const int i         = (hi << 2) + (r2 >> 1);   // VGPR index 0..7
    const int j         = (i << 1) + (r2 & 1);     // element 0..15
    const int lane      = (lane_half << 4) + m16;
    return (lane << 4) + j;
}

// ---------------------------------------------------------------------------
// Generic WMMA GEMM: C[M,N] = epilogue(A[M,K] @ B + bias)
//   TRANSB=false : B is K x N row-major
//   TRANSB=true  : B is N x K row-major (C = A @ B^T)
// Epilogues: 0 none, 1 +bias[n], 2 +bias[n] then exact GELU,
//            3 +rowscale[m]*bias[n]
// 64x64 tile / block, 128 threads (4 waves), K-step 32,
// bf16 hi/lo split -> 3 wmma per 16x16 tile per K-step.
// ---------------------------------------------------------------------------
template <int EP, bool TRANSB>
__global__ __launch_bounds__(128)
void gemm_wmma_kernel(const float* __restrict__ A, const float* __restrict__ Bm,
                      float* __restrict__ C, int M, int N, int K,
                      const float* __restrict__ bias,
                      const float* __restrict__ rowscale)
{
    __shared__ __align__(32) __bf16 Ah[4 * 512];
    __shared__ __align__(32) __bf16 Al[4 * 512];
    __shared__ __align__(32) __bf16 Bh[4 * 512];
    __shared__ __align__(32) __bf16 Bl[4 * 512];

    const int tid  = threadIdx.x;
    const int wave = tid >> 5;
    const int lane = tid & 31;
    const int m0   = blockIdx.y * 64;
    const int n0   = blockIdx.x * 64;

    v8f acc[4];
    for (int i = 0; i < 4; ++i)
        for (int j = 0; j < 8; ++j) acc[i][j] = 0.0f;

    for (int k0 = 0; k0 < K; k0 += 32) {
        // ---- A tile 64x32 -> fragment-order LDS (hi/lo) ----
        for (int v = 0; v < 4; ++v) {
            const int idx4 = tid + v * 128;          // 0..511 float4s
            const int r    = idx4 >> 3;              // row 0..63
            const int kc   = (idx4 & 7) << 2;        // k 0..28
            const float4 f = *(const float4*)(A + (size_t)(m0 + r) * K + k0 + kc);
            const float vals[4] = {f.x, f.y, f.z, f.w};
            const int w = r >> 4, mm = r & 15;
            for (int e = 0; e < 4; ++e) {
                const int fi = w * 512 + frag_idx(mm, kc + e);
                __bf16 h, l; split_bf16(vals[e], h, l);
                Ah[fi] = h; Al[fi] = l;
            }
        }
        // ---- B tile 32x64 (logical KxN) -> fragment-order LDS ----
        if (!TRANSB) {
            for (int v = 0; v < 4; ++v) {
                const int idx4 = tid + v * 128;
                const int kk   = idx4 >> 4;            // 0..31
                const int nc   = (idx4 & 15) << 2;     // 0..60
                const float4 f = *(const float4*)(Bm + (size_t)(k0 + kk) * N + n0 + nc);
                const float vals[4] = {f.x, f.y, f.z, f.w};
                for (int e = 0; e < 4; ++e) {
                    const int n = nc + e;
                    const int fi = (n >> 4) * 512 + frag_idx(n & 15, kk);
                    __bf16 h, l; split_bf16(vals[e], h, l);
                    Bh[fi] = h; Bl[fi] = l;
                }
            }
        } else {
            // logical B[k][n] = Bm[(n0+n)*K + k0+k]
            for (int v = 0; v < 4; ++v) {
                const int idx4 = tid + v * 128;
                const int n    = idx4 >> 3;            // 0..63
                const int kc   = (idx4 & 7) << 2;      // 0..28
                const float4 f = *(const float4*)(Bm + (size_t)(n0 + n) * K + k0 + kc);
                const float vals[4] = {f.x, f.y, f.z, f.w};
                const int nt = n >> 4, nn = n & 15;
                for (int e = 0; e < 4; ++e) {
                    const int fi = nt * 512 + frag_idx(nn, kc + e);
                    __bf16 h, l; split_bf16(vals[e], h, l);
                    Bh[fi] = h; Bl[fi] = l;
                }
            }
        }
        __syncthreads();

        const v16bf ah = *(const v16bf*)(Ah + wave * 512 + lane * 16);
        const v16bf al = *(const v16bf*)(Al + wave * 512 + lane * 16);
        for (int nt = 0; nt < 4; ++nt) {
            const v16bf bh = *(const v16bf*)(Bh + nt * 512 + lane * 16);
            const v16bf bl = *(const v16bf*)(Bl + nt * 512 + lane * 16);
            // a*b ~= ah*bh + ah*bl + al*bh  (al*bl ~2^-16 relative, dropped)
            acc[nt] = __builtin_amdgcn_wmma_f32_16x16x32_bf16(
                          false, ah, false, bh, (short)0, acc[nt], false, false);
            acc[nt] = __builtin_amdgcn_wmma_f32_16x16x32_bf16(
                          false, ah, false, bl, (short)0, acc[nt], false, false);
            acc[nt] = __builtin_amdgcn_wmma_f32_16x16x32_bf16(
                          false, al, false, bh, (short)0, acc[nt], false, false);
        }
        __syncthreads();
    }

    // ---- epilogue + store (C/D layout: VGPR r -> M = r + 8*(lane>=16)) ----
    const int mbase = m0 + wave * 16 + ((lane >> 4) << 3);
    const int nbase = n0 + (lane & 15);
    for (int nt = 0; nt < 4; ++nt) {
        const int n = nbase + nt * 16;
        const float bn = (EP == 0) ? 0.0f : bias[n];
        for (int r = 0; r < 8; ++r) {
            const int m = mbase + r;
            float val = acc[nt][r];
            if (EP == 1 || EP == 2) val += bn;
            if (EP == 3)            val += rowscale[m] * bn;
            if (EP == 2)            val = 0.5f * val * (1.0f + erff(val * 0.70710678118654752f));
            C[(size_t)m * N + n] = val;
        }
    }
}

// ---------------------------------------------------------------------------
// attn logits: logits[bt,n] = (soc_x[bt,n,:]·qk[bt,:] + q[bt,:]·bk) / D
// One block per (b,t); first full streaming pass over soc_x (ascending order).
// Each wave processes 2 neighbor rows at once + prefetches the next pair.
// ---------------------------------------------------------------------------
__global__ __launch_bounds__(256)
void attn_logits_kernel(const float* __restrict__ soc, const float* __restrict__ qk,
                        const float* __restrict__ q,   const float* __restrict__ bk,
                        float* __restrict__ logits)
{
    __shared__ float qks[DD];
    __shared__ float red[256];
    const int bt  = blockIdx.x;
    const int tid = threadIdx.x;

    float p = 0.0f;
    for (int i = tid; i < DD; i += 256) {
        qks[i] = qk[(size_t)bt * DD + i];
        p += q[(size_t)bt * DD + i] * bk[i];
    }
    red[tid] = p;
    __syncthreads();
    for (int s = 128; s > 0; s >>= 1) {
        if (tid < s) red[tid] += red[tid + s];
        __syncthreads();
    }
    const float qbk   = red[0];
    const float inv_d = 1.0f / (float)DD;

    const int wave = tid >> 5, lane = tid & 31;
    for (int ni = 0; ni < 8; ++ni) {
        const int n0 = wave * 2 + ni * 16;       // wave covers {2w, 2w+1} + 16*ni
        const float* sp0 = soc + ((size_t)bt * NN + n0) * DD;
        const float* sp1 = sp0 + DD;
        // speculative prefetch of the next row pair (16 rows ahead)
        __builtin_prefetch(sp0 + 16 * DD + lane * 16, 0, 0);
        __builtin_prefetch(sp1 + 16 * DD + lane * 16, 0, 0);
        float a0 = 0.0f, a1 = 0.0f;
        for (int c = 0; c < 4; ++c) {
            const int d = lane * 4 + c * 128;
            const float4 f0 = *(const float4*)(sp0 + d);
            const float4 f1 = *(const float4*)(sp1 + d);
            a0 += f0.x * qks[d] + f0.y * qks[d + 1] + f0.z * qks[d + 2] + f0.w * qks[d + 3];
            a1 += f1.x * qks[d] + f1.y * qks[d + 1] + f1.z * qks[d + 2] + f1.w * qks[d + 3];
        }
        for (int off = 16; off > 0; off >>= 1) {
            a0 += __shfl_xor(a0, off, 32);
            a1 += __shfl_xor(a1, off, 32);
        }
        if (lane == 0) {
            logits[(size_t)bt * NN + n0]     = (a0 + qbk) * inv_d;
            logits[(size_t)bt * NN + n0 + 1] = (a1 + qbk) * inv_d;
        }
    }
}

// ---------------------------------------------------------------------------
// Per-batch gate: softmax over N, conv gating network (all in LDS),
// sparse_attn = attn*gate -> ws, ssum[bt] = sum_n sparse -> ws.
// Writes softmax attn to the output tensor's attn region. No soc_x access.
// ---------------------------------------------------------------------------
__global__ __launch_bounds__(256)
void gate_kernel(const float* __restrict__ logits,
                 const float* __restrict__ c1w, const float* __restrict__ c1b,
                 const float* __restrict__ c2w, const float* __restrict__ c2b,
                 const float* __restrict__ c3w, const float* __restrict__ c3b,
                 const float* __restrict__ c4w, const float* __restrict__ c4b,
                 float* __restrict__ attn_dst, float* __restrict__ sparse,
                 float* __restrict__ ssum)
{
    __shared__ float att[TT][NN];     // 8 KB
    __shared__ float h1[CHN][NN];     // 32 KB
    __shared__ float h2[CHN][NN];     // 32 KB
    const int b    = blockIdx.x;
    const int tid  = threadIdx.x;
    const int wave = tid >> 5, lane = tid & 31;

    for (int i = tid; i < TT * NN; i += 256)
        att[i >> 7][i & 127] = logits[(size_t)b * TT * NN + i];
    __syncthreads();

    // softmax per row t (wave handles rows wave, wave+8)
    for (int rr = 0; rr < 2; ++rr) {
        const int t = wave + rr * 8;
        float v0 = att[t][lane], v1 = att[t][lane + 32];
        float v2 = att[t][lane + 64], v3 = att[t][lane + 96];
        float mx = fmaxf(fmaxf(v0, v1), fmaxf(v2, v3));
        for (int off = 16; off > 0; off >>= 1) mx = fmaxf(mx, __shfl_xor(mx, off, 32));
        v0 = expf(v0 - mx); v1 = expf(v1 - mx); v2 = expf(v2 - mx); v3 = expf(v3 - mx);
        float s = v0 + v1 + v2 + v3;
        for (int off = 16; off > 0; off >>= 1) s += __shfl_xor(s, off, 32);
        const float inv = 1.0f / s;
        v0 *= inv; v1 *= inv; v2 *= inv; v3 *= inv;
        att[t][lane] = v0; att[t][lane + 32] = v1;
        att[t][lane + 64] = v2; att[t][lane + 96] = v3;
        float* dst = attn_dst + ((size_t)b * TT + t) * NN;
        dst[lane] = v0; dst[lane + 32] = v1; dst[lane + 64] = v2; dst[lane + 96] = v3;
    }
    __syncthreads();

    // h1 = c1w(64x16) @ att + c1b
    for (int i = tid; i < CHN * NN; i += 256) {
        const int oc = i >> 7, n = i & 127;
        float s = c1b[oc];
        for (int k = 0; k < TT; ++k) s += c1w[oc * TT + k] * att[k][n];
        h1[oc][n] = s;
    }
    __syncthreads();
    // h2 = relu(c2w(64x64) @ h1 + c2b)
    for (int i = tid; i < CHN * NN; i += 256) {
        const int oc = i >> 7, n = i & 127;
        float s = c2b[oc];
        for (int k = 0; k < CHN; ++k) s += c2w[oc * CHN + k] * h1[k][n];
        h2[oc][n] = fmaxf(s, 0.0f);
    }
    __syncthreads();
    // h3 = c3w @ h2 + c3b  (staged through registers, stored back into h1)
    float tmp[32];
    for (int v = 0; v < 32; ++v) {
        const int i = tid + v * 256;
        const int oc = i >> 7, n = i & 127;
        float s = c3b[oc];
        for (int k = 0; k < CHN; ++k) s += c3w[oc * CHN + k] * h2[k][n];
        tmp[v] = s;
    }
    __syncthreads();
    for (int v = 0; v < 32; ++v) {
        const int i = tid + v * 256;
        h1[i >> 7][i & 127] = tmp[v];
    }
    __syncthreads();
    // gate = sigmoid(c4w(16x64) @ h3 + c4b); threshold; sparse = att * gate
    for (int i = tid; i < TT * NN; i += 256) {
        const int t = i >> 7, n = i & 127;
        float s = c4b[t];
        for (int k = 0; k < CHN; ++k) s += c4w[t * CHN + k] * h1[k][n];
        float g = 1.0f / (1.0f + expf(-s));
        if (g < KESI) g = 0.0f;
        const float sp = att[t][n] * g;
        att[t][n] = sp;
        sparse[(size_t)(b * TT + t) * NN + n] = sp;
    }
    __syncthreads();

    // ssum per row
    for (int rr = 0; rr < 2; ++rr) {
        const int t = wave + rr * 8;
        float s = att[t][lane] + att[t][lane + 32] + att[t][lane + 64] + att[t][lane + 96];
        for (int off = 16; off > 0; off >>= 1) s += __shfl_xor(s, off, 32);
        if (lane == 0) ssum[b * TT + t] = s;
    }
}

// ---------------------------------------------------------------------------
// sx[bt,d] = sum_n sparse[bt,n] * soc_x[bt,n,d]   (2nd pass over soc_x)
// 1024 blocks traversed in REVERSE block order: the logits kernel streamed
// soc_x ascending, so the tail (~192 MB = L2 size) is still resident; reading
// it back MRU-first converts most of this pass into L2 hits.
// 256 threads: 2 n-halves x 128 d-quarters (float4), n unrolled 4-deep for
// enough loads in flight to saturate HBM.
// ---------------------------------------------------------------------------
__global__ __launch_bounds__(256)
void sx_kernel(const float* __restrict__ soc, const float* __restrict__ sparse,
               float* __restrict__ sx)
{
    __shared__ float sws[NN];
    __shared__ float red1[DD];
    const int bt   = (BT - 1) - blockIdx.x;       // reverse traversal
    const int tid  = threadIdx.x;
    const int half = tid >> 7;                    // n-half: 0 -> [0,64), 1 -> [64,128)
    const int d    = (tid & 127) * 4;

    if (tid < NN) sws[tid] = sparse[(size_t)bt * NN + tid];
    __syncthreads();

    const float* sp = soc + (size_t)bt * NN * DD;
    float4 a0 = {0,0,0,0}, a1 = {0,0,0,0}, a2 = {0,0,0,0}, a3 = {0,0,0,0};
    for (int i = 60; i >= 0; i -= 4) {            // descending n within half
        const int n = half * 64 + i;
        const float w0 = sws[n], w1 = sws[n + 1], w2 = sws[n + 2], w3 = sws[n + 3];
        const float4 f0 = *(const float4*)(sp + (size_t)n * DD + d);
        const float4 f1 = *(const float4*)(sp + (size_t)(n + 1) * DD + d);
        const float4 f2 = *(const float4*)(sp + (size_t)(n + 2) * DD + d);
        const float4 f3 = *(const float4*)(sp + (size_t)(n + 3) * DD + d);
        a0.x += w0 * f0.x; a0.y += w0 * f0.y; a0.z += w0 * f0.z; a0.w += w0 * f0.w;
        a1.x += w1 * f1.x; a1.y += w1 * f1.y; a1.z += w1 * f1.z; a1.w += w1 * f1.w;
        a2.x += w2 * f2.x; a2.y += w2 * f2.y; a2.z += w2 * f2.z; a2.w += w2 * f2.w;
        a3.x += w3 * f3.x; a3.y += w3 * f3.y; a3.z += w3 * f3.z; a3.w += w3 * f3.w;
    }
    float4 a;
    a.x = (a0.x + a1.x) + (a2.x + a3.x);
    a.y = (a0.y + a1.y) + (a2.y + a3.y);
    a.z = (a0.z + a1.z) + (a2.z + a3.z);
    a.w = (a0.w + a1.w) + (a2.w + a3.w);

    if (half == 1) {
        red1[d] = a.x; red1[d + 1] = a.y; red1[d + 2] = a.z; red1[d + 3] = a.w;
    }
    __syncthreads();
    if (half == 0) {
        float4 r;
        r.x = a.x + red1[d];     r.y = a.y + red1[d + 1];
        r.z = a.z + red1[d + 2]; r.w = a.w + red1[d + 3];
        *(float4*)(sx + (size_t)bt * DD + d) = r;
    }
}

// ---------------------------------------------------------------------------
// out[row] = LayerNorm(a[row] + res[row]) * g + beta  (D = 512, eps 1e-5)
// ---------------------------------------------------------------------------
__global__ __launch_bounds__(256)
void ln_residual_kernel(const float* __restrict__ a, const float* __restrict__ res,
                        const float* __restrict__ g, const float* __restrict__ beta,
                        float* __restrict__ out)
{
    __shared__ float red[256];
    const int row = blockIdx.x, tid = threadIdx.x;
    const size_t base = (size_t)row * DD;
    const float z0 = a[base + tid]       + res[base + tid];
    const float z1 = a[base + tid + 256] + res[base + tid + 256];

    red[tid] = z0 + z1;
    __syncthreads();
    for (int s = 128; s > 0; s >>= 1) { if (tid < s) red[tid] += red[tid + s]; __syncthreads(); }
    const float mean = red[0] * (1.0f / (float)DD);
    __syncthreads();

    const float d0 = z0 - mean, d1 = z1 - mean;
    red[tid] = d0 * d0 + d1 * d1;
    __syncthreads();
    for (int s = 128; s > 0; s >>= 1) { if (tid < s) red[tid] += red[tid + s]; __syncthreads(); }
    const float rstd = rsqrtf(red[0] * (1.0f / (float)DD) + 1e-5f);

    out[base + tid]       = d0 * rstd * g[tid]       + beta[tid];
    out[base + tid + 256] = d1 * rstd * g[tid + 256] + beta[tid + 256];
}

// ---------------------------------------------------------------------------
// Host orchestration
// ---------------------------------------------------------------------------
extern "C" void kernel_launch(void* const* d_in, const int* in_sizes, int n_in,
                              void* d_out, int out_size, void* d_ws, size_t ws_size,
                              hipStream_t stream)
{
    (void)in_sizes; (void)n_in; (void)out_size; (void)ws_size;

    const float* x    = (const float*)d_in[0];
    const float* soc  = (const float*)d_in[1];
    const float* wq   = (const float*)d_in[2];
    const float* bq   = (const float*)d_in[3];
    const float* wk   = (const float*)d_in[4];
    const float* bk   = (const float*)d_in[5];
    const float* wv   = (const float*)d_in[6];
    const float* bv   = (const float*)d_in[7];
    const float* wo   = (const float*)d_in[8];
    const float* bo   = (const float*)d_in[9];
    const float* c1w  = (const float*)d_in[10];
    const float* c1b  = (const float*)d_in[11];
    const float* c2w  = (const float*)d_in[12];
    const float* c2b  = (const float*)d_in[13];
    const float* c3w  = (const float*)d_in[14];
    const float* c3b  = (const float*)d_in[15];
    const float* c4w  = (const float*)d_in[16];
    const float* c4b  = (const float*)d_in[17];
    const float* ln1g = (const float*)d_in[18];
    const float* ln1b = (const float*)d_in[19];
    const float* ln2g = (const float*)d_in[20];
    const float* ln2b = (const float*)d_in[21];
    const float* fw1  = (const float*)d_in[22];
    const float* fb1  = (const float*)d_in[23];
    const float* fw2  = (const float*)d_in[24];
    const float* fb2  = (const float*)d_in[25];

    float* out       = (float*)d_out;                 // (B,T,D)
    float* attn_out  = out + (size_t)BT * DD;         // (B,T,1,N)

    // workspace layout (floats); overlapped by lifetime:
    //   [0 .. 2359296) : q | qk | logits | sparse | svw | aout (dead pre-FFN1)
    //   ffn1 reuses [0 .. 2097152)
    float* ws     = (float*)d_ws;
    float* q      = ws;                               //  512K
    float* qk     = ws + 524288;                      //  512K
    float* logits = ws + 1048576;                     //  128K
    float* sparse = ws + 1179648;                     //  128K
    float* svw    = ws + 1310720;                     //  512K
    float* aout   = ws + 1835008;                     //  512K  (ends 2359296)
    float* ffn1   = ws;                               // 2048K  (reuse; safe)
    float* sx     = ws + 2359296;                     //  512K
    float* ssum   = ws + 2883584;                     //    1K
    float* x1     = ws + 2884608;                     //  512K
    float* ffn2   = ws + 3408896;                     //  512K  (total ~15.7 MB)

    const dim3 blk(128);
    const dim3 g512(EE / 64, BT / 64);     // (8,16)
    const dim3 gff1(FFD / 64, BT / 64);    // (32,16)

    // 1) q = x @ wq + bq
    gemm_wmma_kernel<1, false><<<g512, blk, 0, stream>>>(x, wq, q, BT, EE, DD, bq, nullptr);
    // 2) qk = q @ wk^T
    gemm_wmma_kernel<0, true><<<g512, blk, 0, stream>>>(q, wk, qk, BT, DD, EE, nullptr, nullptr);
    // 3) attention logits (1st pass over soc_x, ascending)
    attn_logits_kernel<<<dim3(BT), dim3(256), 0, stream>>>(soc, qk, q, bk, logits);
    // 4) softmax + conv gate + sparsify (no soc_x); writes attn output, sparse, ssum
    gate_kernel<<<dim3(BB), dim3(256), 0, stream>>>(logits,
        c1w, c1b, c2w, c2b, c3w, c3b, c4w, c4b, attn_out, sparse, ssum);
    // 5) sx = sparse-weighted reduction of soc_x (2nd pass, reverse for L2 reuse)
    sx_kernel<<<dim3(BT), dim3(256), 0, stream>>>(soc, sparse, sx);
    // 6) svw = sx @ wv + ssum*bv
    gemm_wmma_kernel<3, false><<<g512, blk, 0, stream>>>(sx, wv, svw, BT, EE, DD, bv, ssum);
    // 7) aout = svw @ wo + bo
    gemm_wmma_kernel<1, false><<<g512, blk, 0, stream>>>(svw, wo, aout, BT, DD, EE, bo, nullptr);
    // 8) x1 = LN(aout + x)
    ln_residual_kernel<<<dim3(BT), dim3(256), 0, stream>>>(aout, x, ln1g, ln1b, x1);
    // 9) ffn1 = gelu(x1 @ fw1 + fb1)
    gemm_wmma_kernel<2, false><<<gff1, blk, 0, stream>>>(x1, fw1, ffn1, BT, FFD, DD, fb1, nullptr);
    // 10) ffn2 = ffn1 @ fw2 + fb2
    gemm_wmma_kernel<1, false><<<g512, blk, 0, stream>>>(ffn1, fw2, ffn2, BT, DD, FFD, fb2, nullptr);
    // 11) out = LN(ffn2 + x1)
    ln_residual_kernel<<<dim3(BT), dim3(256), 0, stream>>>(ffn2, x1, ln2g, ln2b, out);
}
